// gen_heter_Emb_p_58987080843449
// MI455X (gfx1250) — matrix operand order repaired and checked by reading, not compiled, not run on hardware
//
#include <hip/hip_runtime.h>
#include <hip/hip_bf16.h>

typedef __attribute__((ext_vector_type(16))) _Float16 v16h;
typedef __attribute__((ext_vector_type(8)))  _Float16 v8h;
typedef __attribute__((ext_vector_type(4)))  _Float16 v4h;
typedef __attribute__((ext_vector_type(8)))  float    v8f;
typedef __attribute__((ext_vector_type(4)))  unsigned int v4u;
typedef __attribute__((ext_vector_type(8)))  int      v8i;
typedef __attribute__((ext_vector_type(4)))  int      v4i;

#define BM 64   // rows per workgroup (4 WMMA M-tiles)
#define NW 512  // all MLP layer widths are 512

#if __has_builtin(__builtin_amdgcn_tensor_load_to_lds) && \
    __has_builtin(__builtin_amdgcn_s_wait_tensorcnt)
#define HAVE_TDM 1
#else
#define HAVE_TDM 0
#endif

// ---------------------------------------------------------------------------
// fp32 -> fp16 elementwise convert (vectorized by 4)
// ---------------------------------------------------------------------------
__global__ void cvt_f32_f16(const float* __restrict__ src,
                            _Float16* __restrict__ dst, int n4) {
  int i = blockIdx.x * blockDim.x + threadIdx.x;
  if (i < n4) {
    float4 v = ((const float4*)src)[i];
    v4h o;
    o[0] = (_Float16)v.x; o[1] = (_Float16)v.y;
    o[2] = (_Float16)v.z; o[3] = (_Float16)v.w;
    ((v4h*)dst)[i] = o;
  }
}

// ---------------------------------------------------------------------------
// fp32 (K x N) -> f16 transposed (N x K), 32x32 tiles via LDS.
// Done once per weight so the GEMM hot loop needs no transpose.
// ---------------------------------------------------------------------------
__global__ void __launch_bounds__(256) cvt_transpose_f32_f16(
    const float* __restrict__ src, _Float16* __restrict__ dst, int K, int N) {
  __shared__ float tile[32][33];
  const int kb = blockIdx.x * 32;
  const int nb = blockIdx.y * 32;
  const int tx = threadIdx.x & 31;
  const int ty = threadIdx.x >> 5;          // 0..7
  #pragma unroll
  for (int r = ty; r < 32; r += 8)
    tile[r][tx] = src[(size_t)(kb + r) * N + nb + tx];
  __syncthreads();
  #pragma unroll
  for (int r = ty; r < 32; r += 8)
    dst[(size_t)(nb + r) * K + kb + tx] = (_Float16)tile[tx][r];
}

// ---------------------------------------------------------------------------
// Positional FCN: per-edge 5 -> 64 (relu) -> 64 (relu) -> 128 (linear)
// ---------------------------------------------------------------------------
__global__ void __launch_bounds__(256) pos_mlp_kernel(
    const float* __restrict__ centroid,
    const float* __restrict__ w1, const float* __restrict__ b1,
    const float* __restrict__ w2, const float* __restrict__ b2,
    const float* __restrict__ w3, const float* __restrict__ b3,
    _Float16* __restrict__ posF16, int n, int E)
{
  __shared__ float sw1[5 * 64], sb1[64], sw2[64 * 64], sb2[64];
  __shared__ float sw3[64 * 128], sb3[128];
  for (int i = threadIdx.x; i < 5 * 64;  i += 256) sw1[i] = w1[i];
  for (int i = threadIdx.x; i < 64;      i += 256) { sb1[i] = b1[i]; sb2[i] = b2[i]; }
  for (int i = threadIdx.x; i < 64 * 64; i += 256) sw2[i] = w2[i];
  for (int i = threadIdx.x; i < 64 * 128;i += 256) sw3[i] = w3[i];
  for (int i = threadIdx.x; i < 128;     i += 256) sb3[i] = b3[i];
  __syncthreads();

  const int nm1 = n - 1;
  for (int e = blockIdx.x * blockDim.x + threadIdx.x; e < E;
       e += gridDim.x * blockDim.x) {
    int i = e / nm1;
    int r = e - i * nm1;
    int j = r + (r >= i ? 1 : 0);
    float dx = centroid[i * 3 + 0] - centroid[j * 3 + 0];
    float dy = centroid[i * 3 + 1] - centroid[j * 3 + 1];
    float dist = sqrtf(dx * dx + dy * dy);
    float inv = 1.0f / dist;
    float f[5] = {dx, dy, dist, dx * inv, dy * inv};

    float h1[64];
    for (int u = 0; u < 64; ++u) {
      float s = sb1[u];
      #pragma unroll
      for (int k = 0; k < 5; ++k) s = fmaf(f[k], sw1[k * 64 + u], s);
      h1[u] = fmaxf(s, 0.0f);
    }
    float h2[64];
    for (int u = 0; u < 64; ++u) {
      float s = sb2[u];
      for (int k = 0; k < 64; ++k) s = fmaf(h1[k], sw2[k * 64 + u], s);
      h2[u] = fmaxf(s, 0.0f);
    }
    for (int o = 0; o < 128; ++o) {
      float s = sb3[o];
      for (int k = 0; k < 64; ++k) s = fmaf(h2[k], sw3[k * 128 + o], s);
      posF16[(size_t)e * 128 + o] = (_Float16)s;   // linear (no relu)
    }
  }
}

// ---------------------------------------------------------------------------
// Tensor Data Mover: 2D tile (tile_k halves x tile_rows, row stride
// stride_halves) from global f16 -> contiguous LDS at lds_byte_off.
// D# built per CDNA5 ISA ch.8 (group0: count/lds/global/type=2; group1:
// data_size=2B, tensor dims, tile dims, dim0 stride). Issued by one wave,
// completion via TENSORcnt.
// ---------------------------------------------------------------------------
#if HAVE_TDM
__device__ __forceinline__ void tdm_load_2d(unsigned lds_byte_off,
                                            const _Float16* gsrc,
                                            unsigned tile_k,       // halves/row
                                            unsigned tile_rows,
                                            unsigned stride_halves)
{
  unsigned long long ga = (unsigned long long)(uintptr_t)gsrc;
  v4u g0 = (v4u)0u;
  g0[0] = 1u;                                       // count=1 (valid), user D#
  g0[1] = lds_byte_off;                             // lds_addr (bytes)
  g0[2] = (unsigned)(ga & 0xFFFFFFFFu);             // global_addr[31:0]
  g0[3] = (unsigned)((ga >> 32) & 0x01FFFFFFu)      // global_addr[56:32]
          | 0x80000000u;                            // type=2 ("image")
  v8i g1 = (v8i)0;
  g1[0] = (int)(1u << 16);                          // data_size=1 -> 2 bytes
  g1[1] = (int)((stride_halves & 0xFFFFu) << 16);   // tensor_dim0[15:0]
  g1[2] = (int)((stride_halves >> 16) |
                ((tile_rows & 0xFFFFu) << 16));     // dim0 hi | tensor_dim1 lo
  g1[3] = (int)((tile_rows >> 16) |
                ((tile_k & 0xFFFFu) << 16));        // dim1 hi | tile_dim0
  g1[4] = (int)(tile_rows & 0xFFFFu);               // tile_dim1 (tile_dim2=0)
  g1[5] = (int)stride_halves;                       // tensor_dim0_stride lo32
  g1[6] = 0;
  g1[7] = 0;
  v4i g2 = (v4i)0;
  v4i g3 = (v4i)0;
#if defined(__clang_major__) && (__clang_major__ >= 23)
  v8i gx = (v8i)0;
  __builtin_amdgcn_tensor_load_to_lds(g0, g1, g2, g3, gx, 0);
#else
  __builtin_amdgcn_tensor_load_to_lds(g0, g1, g2, g3, 0);
#endif
}
#endif

// ---------------------------------------------------------------------------
// WMMA fragment loaders (ISA 7.12.2 wave32 layouts, 16x16x32 f16)
// ---------------------------------------------------------------------------
__device__ __forceinline__ v16h a_frag_lds(const _Float16* __restrict__ row,
                                           int k0, int lane) {
  const int base = (lane & 16) ? 8 : 0;
  v16h a;
  #pragma unroll
  for (int j = 0; j < 4; ++j) {
    a[2 * j]     = row[k0 + base + 2 * j];
    a[2 * j + 1] = row[k0 + base + 2 * j + 1];
  }
  #pragma unroll
  for (int j = 0; j < 4; ++j) {
    a[8 + 2 * j]     = row[k0 + 16 + base + 2 * j];
    a[8 + 2 * j + 1] = row[k0 + 16 + base + 2 * j + 1];
  }
  return a;
}

// B panel in LDS is n-major: Bp[n*32 + k]
__device__ __forceinline__ v16h b_frag_lds(const _Float16* __restrict__ Bp,
                                           int ncol, int lane) {
  const int kb = (lane & 16) ? 8 : 0;
  const _Float16* p = Bp + ncol * 32 + kb;
  v16h b;
  #pragma unroll
  for (int j = 0; j < 8; ++j) {
    b[2 * j]     = p[2 * j + (j >= 4 ? 8 : 0)];
    b[2 * j + 1] = p[2 * j + 1 + (j >= 4 ? 8 : 0)];
  }
  return b;
}

__device__ __forceinline__ v8f v8f_zero() {
  v8f z;
  #pragma unroll
  for (int q = 0; q < 8; ++q) z[q] = 0.0f;
  return z;
}

// LDS layout (halves): Xb [64*512] | Hb [64*512] | Bp [512*32]
#define XB_OFF 0
#define HB_OFF (64 * NW)
#define BP_OFF (2 * 64 * NW)

// Load one 32-wide k-panel of transposed weights WT (N x Ktot) into Bp.
__device__ __forceinline__ void load_b_panel(_Float16* __restrict__ Bp,
                                             const _Float16* __restrict__ WT,
                                             int Ktot, int k0,
                                             int tid, int wave) {
#if HAVE_TDM
  if (wave == 0) {
    tdm_load_2d((unsigned)(BP_OFF * sizeof(_Float16)),
                WT + k0, /*tile_k=*/32, /*rows=*/NW,
                /*stride=*/(unsigned)Ktot);
    __builtin_amdgcn_s_wait_tensorcnt((short)0);
  }
#else
  #pragma unroll 1
  for (int idx = tid; idx < NW * 4; idx += 256) {   // 512 rows x 4 v8h chunks
    int nrow = idx >> 2;
    int c8   = (idx & 3) << 3;
    *(v8h*)(Bp + nrow * 32 + c8) =
        *(const v8h*)(WT + (size_t)nrow * Ktot + k0 + c8);
  }
#endif
}

// ---------------------------------------------------------------------------
// Fused two-layer MLP: out = relu(relu(X @ W1 + b1) @ W2 + b2)
// X = up to 3 row-major f16 segments (concat along K). Weights pre-transposed
// f16: W1T (512 x K1tot), W2T (512 x 512). Hidden (64x512 f16) stays in LDS.
// 8 waves; wave (mtile, nhalf) owns 16 rows x 256 cols -> 16 v8f accums.
// ---------------------------------------------------------------------------
__global__ void __launch_bounds__(256) mlp2_wmma_kernel(
    const _Float16* __restrict__ seg0, int w0,
    const _Float16* __restrict__ seg1, int w1n,
    const _Float16* __restrict__ seg2, int w2n,
    const _Float16* __restrict__ W1T, const float* __restrict__ B1,
    const _Float16* __restrict__ W2T, const float* __restrict__ B2,
    float* __restrict__ outF32, _Float16* __restrict__ outF16)
{
  extern __shared__ _Float16 smem[];
  _Float16* Xb = smem + XB_OFF;        // [64][segW] contiguous per segment
  _Float16* Hb = smem + HB_OFF;        // [64][512]
  _Float16* Bp = smem + BP_OFF;        // [512][32] n-major panel

  const int tid   = threadIdx.x;
  const int lane  = tid & 31;
  const int wave  = tid >> 5;
  const int mtile = wave >> 1;          // 0..3  -> rows 16*mtile..+15
  const int nhalf = wave & 1;           // 0..1  -> cols nhalf*256..+255
  const int row0  = blockIdx.x * BM;
  const int rlane = lane & 15;
  const int rloc  = 16 * mtile + rlane;
  const int ncol0 = nhalf * 256;
  const int mlane = (lane & 16) ? 8 : 0;   // C/D: VGPR r -> M = r (+8 hi lanes)

  const int Ktot = w0 + w1n + w2n;

  v8f acc[16];
  #pragma unroll
  for (int t = 0; t < 16; ++t) acc[t] = v8f_zero();

  const _Float16* segs[3] = {seg0, seg1, seg2};
  const int       segw[3] = {w0, w1n, w2n};

  // ----------------- Layer 1 -----------------
  int segoff = 0;
  for (int s = 0; s < 3; ++s) {
    const int W = segw[s];
    if (W == 0) continue;
    const _Float16* S = segs[s];

    // X tile (64 x W, contiguous, row stride W) -> Xb
#if HAVE_TDM
    if (wave == 0) {
      tdm_load_2d((unsigned)(XB_OFF * sizeof(_Float16)),
                  S + (size_t)row0 * W, /*tile_k=*/(unsigned)W, /*rows=*/64,
                  /*stride=*/(unsigned)W);
      __builtin_amdgcn_s_wait_tensorcnt((short)0);
    }
#else
    {
      const int wdiv8 = W >> 3;
      #pragma unroll 1
      for (int idx = tid; idx < 64 * wdiv8; idx += 256) {
        int r  = idx / wdiv8;
        int c8 = (idx - r * wdiv8) << 3;
        *(v8h*)(Xb + r * W + c8) =
            *(const v8h*)(S + (size_t)(row0 + r) * W + c8);
      }
    }
#endif
    __syncthreads();

    for (int k0 = 0; k0 < W; k0 += 32) {
      load_b_panel(Bp, W1T, Ktot, segoff + k0, tid, wave);
      if (k0 + 32 < W)                // emits global_prefetch_b8
        __builtin_prefetch(W1T + (size_t)(segoff + k0 + 32), 0, 1);
      __syncthreads();

      v16h a = a_frag_lds(Xb + rloc * W, k0, lane);
      #pragma unroll
      for (int t = 0; t < 16; ++t) {
        v16h b = b_frag_lds(Bp, ncol0 + t * 16 + rlane, lane);
        acc[t] = __builtin_amdgcn_wmma_f32_16x16x32_f16(
            false, a, false, b, (short)0, acc[t], false, false);
      }
      __syncthreads();
    }
    segoff += W;
  }

  // epilogue layer 1: bias + relu -> Hb (f16)
  #pragma unroll
  for (int t = 0; t < 16; ++t) {
    const int n = ncol0 + t * 16 + rlane;
    const float bias = B1[n];
    #pragma unroll
    for (int r8 = 0; r8 < 8; ++r8) {
      float v = fmaxf(acc[t][r8] + bias, 0.0f);
      Hb[(16 * mtile + mlane + r8) * NW + n] = (_Float16)v;
    }
  }
  __syncthreads();

  // ----------------- Layer 2 (K = 512, A = Hb) -----------------
  #pragma unroll
  for (int t = 0; t < 16; ++t) acc[t] = v8f_zero();

  for (int k0 = 0; k0 < NW; k0 += 32) {
    load_b_panel(Bp, W2T, NW, k0, tid, wave);
    if (k0 + 32 < NW)
      __builtin_prefetch(W2T + (size_t)(k0 + 32), 0, 1);
    __syncthreads();

    v16h a = a_frag_lds(Hb + rloc * NW, k0, lane);
    #pragma unroll
    for (int t = 0; t < 16; ++t) {
      v16h b = b_frag_lds(Bp, ncol0 + t * 16 + rlane, lane);
      acc[t] = __builtin_amdgcn_wmma_f32_16x16x32_f16(
          false, a, false, b, (short)0, acc[t], false, false);
    }
    __syncthreads();
  }

  // epilogue layer 2: bias + relu -> fp32 output (+ optional f16 copy)
  #pragma unroll
  for (int t = 0; t < 16; ++t) {
    const int n = ncol0 + t * 16 + rlane;
    const float bias = B2[n];
    #pragma unroll
    for (int r8 = 0; r8 < 8; ++r8) {
      float v = fmaxf(acc[t][r8] + bias, 0.0f);
      size_t gi = (size_t)(row0 + 16 * mtile + mlane + r8) * NW + n;
      outF32[gi] = v;
      if (outF16) outF16[gi] = (_Float16)v;
    }
  }
}

// ---------------------------------------------------------------------------
// Host orchestration
// ---------------------------------------------------------------------------
extern "C" void kernel_launch(void* const* d_in, const int* in_sizes, int n_in,
                              void* d_out, int out_size, void* d_ws, size_t ws_size,
                              hipStream_t stream)
{
  const float* pred    = (const float*)d_in[1];
  const float* centro  = (const float*)d_in[4];
  const float* s_w1 = (const float*)d_in[5];  const float* s_b1 = (const float*)d_in[6];
  const float* s_w2 = (const float*)d_in[7];  const float* s_b2 = (const float*)d_in[8];
  const float* p_w1 = (const float*)d_in[9];  const float* p_b1 = (const float*)d_in[10];
  const float* p_w2 = (const float*)d_in[11]; const float* p_b2 = (const float*)d_in[12];
  const float* c_w1 = (const float*)d_in[13]; const float* c_b1 = (const float*)d_in[14];
  const float* c_w2 = (const float*)d_in[15]; const float* c_b2 = (const float*)d_in[16];
  const float* l_w1 = (const float*)d_in[17]; const float* l_b1 = (const float*)d_in[18];
  const float* l_w2 = (const float*)d_in[19]; const float* l_b2 = (const float*)d_in[20];
  const float* pw1 = (const float*)d_in[21];  const float* pb1 = (const float*)d_in[22];
  const float* pw2 = (const float*)d_in[23];  const float* pb2 = (const float*)d_in[24];
  const float* pw3 = (const float*)d_in[25];  const float* pb3 = (const float*)d_in[26];

  const int n = in_sizes[4] / 3;        // 256
  const int E = n * (n - 1);            // 65280 (divisible by 64)
  const size_t Es = (size_t)E;

  // workspace layout (f16 halves)
  _Float16* wsh = (_Float16*)d_ws;
  size_t off = 0;
  auto take = [&](size_t cnt) {
    _Float16* p = wsh + off;
    off += (cnt + 127) & ~(size_t)127;
    return p;
  };
  _Float16* predF = take(Es * 512);
  _Float16* posF  = take(Es * 128);
  _Float16* supF  = take(Es * 512);
  _Float16* proxF = take(Es * 512);
  _Float16* compF = take(Es * 512);
  _Float16* wS1 = take(512 * 512);  _Float16* wS2 = take(512 * 512);
  _Float16* wP1 = take(640 * 512);  _Float16* wP2 = take(512 * 512);
  _Float16* wC1 = take(512 * 512);  _Float16* wC2 = take(512 * 512);
  _Float16* wL1 = take((size_t)1536 * 512);  _Float16* wL2 = take(512 * 512);
  (void)ws_size;

  // plain convert for activations
  {
    int n4 = (int)(Es * 512 / 4);
    cvt_f32_f16<<<(n4 + 255) / 256, 256, 0, stream>>>(pred, predF, n4);
  }
  // convert + transpose weights: (K x 512) fp32 -> (512 x K) f16
  auto cvtT = [&](const float* src, _Float16* dst, int K) {
    dim3 g(K / 32, 512 / 32);
    cvt_transpose_f32_f16<<<g, 256, 0, stream>>>(src, dst, K, 512);
  };
  cvtT(s_w1, wS1, 512);   cvtT(s_w2, wS2, 512);
  cvtT(p_w1, wP1, 640);   cvtT(p_w2, wP2, 512);
  cvtT(c_w1, wC1, 512);   cvtT(c_w2, wC2, 512);
  cvtT(l_w1, wL1, 1536);  cvtT(l_w2, wL2, 512);

  pos_mlp_kernel<<<512, 256, 0, stream>>>(centro, pw1, pb1, pw2, pb2, pw3, pb3,
                                          posF, n, E);

  // output slices: tuple order (link, support, prox, comp)
  float* outLink = (float*)d_out;
  float* outSup  = outLink + Es * 512;
  float* outProx = outLink + 2 * Es * 512;
  float* outComp = outLink + 3 * Es * 512;

  const size_t SMEM = (size_t)(64 * NW + 64 * NW + NW * 32) * sizeof(_Float16); // 160 KB
  (void)hipFuncSetAttribute(reinterpret_cast<const void*>(mlp2_wmma_kernel),
                            hipFuncAttributeMaxDynamicSharedMemorySize,
                            (int)SMEM);
  const int gemm_blocks = E / BM;       // 1020

  mlp2_wmma_kernel<<<gemm_blocks, 256, SMEM, stream>>>(
      predF, 512, nullptr, 0, nullptr, 0,
      wS1, s_b1, wS2, s_b2, outSup, supF);
  mlp2_wmma_kernel<<<gemm_blocks, 256, SMEM, stream>>>(
      predF, 512, nullptr, 0, nullptr, 0,
      wC1, c_b1, wC2, c_b2, outComp, compF);
  mlp2_wmma_kernel<<<gemm_blocks, 256, SMEM, stream>>>(
      predF, 512, posF, 128, nullptr, 0,
      wP1, p_b1, wP2, p_b2, outProx, proxF);
  mlp2_wmma_kernel<<<gemm_blocks, 256, SMEM, stream>>>(
      supF, 512, proxF, 512, compF, 512,
      wL1, l_b1, wL2, l_b2, outLink, nullptr);
}
